// DeformConvBlock_36361193128275
// MI455X (gfx1250) — matrix operand order repaired and checked by reading, not compiled, not run on hardware
//
#include <hip/hip_runtime.h>
#include <hip/hip_bf16.h>

typedef __attribute__((ext_vector_type(16))) __bf16 v16bf;
typedef __attribute__((ext_vector_type(8)))  __bf16 v8bf;
typedef __attribute__((ext_vector_type(8)))  float  v8f;

union FragBF { v16bf v; v8bf h[2]; };

#define BATCH 16
#define CIN   192
#define COUT  384
#define HH    64
#define WW    64
#define HW    4096        // 64*64
#define NPIX  65536       // BATCH*HW
#define XPITCH 4356       // 66*66 padded plane (1-pixel zero halo)
#define KT_OFF 54         // 9 taps * (192/32) c-tiles
#define KT_PW  6          // 192/32

__device__ __forceinline__ v8f wmma_bf16(v16bf a, v16bf b, v8f c) {
  return __builtin_amdgcn_wmma_f32_16x16x32_bf16(false, a, false, b, (short)0, c, false, false);
}

// ---------------------------------------------------------------------------
// k_zero16: zero-fill the padded NHWC buffer (halo must read as 0)
// ---------------------------------------------------------------------------
__global__ __launch_bounds__(256) void k_zero16(v8bf* __restrict__ p, int n) {
  int i = blockIdx.x * 256 + threadIdx.x;
  if (i < n) {
    v8bf z;
#pragma unroll
    for (int e = 0; e < 8; ++e) z[e] = (__bf16)0.0f;
    p[i] = z;
  }
}

// ---------------------------------------------------------------------------
// k0a: x f32 NCHW -> bf16 padded NHWC, LDS-tiled 32x32 transpose
// grid: 16(b) * 6(c-tile) * 128(hw-tile) blocks of 256
// ---------------------------------------------------------------------------
__global__ __launch_bounds__(256) void k_x_to_nhwc(const float* __restrict__ x,
                                                   __bf16* __restrict__ xTp) {
  __shared__ float tile[32][33];
  int blk = blockIdx.x;
  int ht = blk & 127; blk >>= 7;
  int ct = blk % 6;   int b = blk / 6;
  int tx = threadIdx.x & 31;
  int ty = threadIdx.x >> 5;       // 0..7
  int hw0 = ht << 5, c0 = ct << 5;
#pragma unroll
  for (int i = 0; i < 4; ++i) {
    int c = c0 + ty + (i << 3);
    tile[ty + (i << 3)][tx] = x[(((size_t)b * CIN + c) << 12) + hw0 + tx];
  }
  __syncthreads();
#pragma unroll
  for (int i = 0; i < 4; ++i) {
    int hw = hw0 + ty + (i << 3);
    int y = hw >> 6, xq = hw & 63;
    xTp[((size_t)(b * XPITCH + (y + 1) * 66 + (xq + 1))) * CIN + c0 + tx] =
        (__bf16)tile[tx][ty + (i << 3)];
  }
}

// ---------------------------------------------------------------------------
// k0b: pack w_off [18,192,3,3] into WMMA B-fragment table
// layout: [kt(54)][nt(2)][lane(32)][i(16)] bf16, K order = tap*192 + c
// ---------------------------------------------------------------------------
__global__ __launch_bounds__(256) void k_pack_woff(const float* __restrict__ w_off,
                                                   __bf16* __restrict__ Bf) {
  int idx = blockIdx.x * 256 + threadIdx.x;
  if (idx >= KT_OFF * 2 * 512) return;
  int i    = idx & 15;
  int lane = (idx >> 4) & 31;
  int nt   = (idx >> 9) & 1;
  int kt   = idx >> 10;
  int col = lane & 15;
  int klocal = ((lane >> 4) << 3) + (i & 7) + ((i >> 3) << 4);       // 0..31 within K-tile
  int kk = kt / 6, ct = kt - kk * 6;
  int ki = kk / 3, kj = kk - ki * 3;
  int c = (ct << 5) + klocal;
  int o = (nt << 4) + col;
  float v = (o < 18) ? w_off[((((size_t)o * CIN) + c) * 3 + ki) * 3 + kj] : 0.0f;
  Bf[idx] = (__bf16)v;
}

// ---------------------------------------------------------------------------
// k0c: pack w_pw [384,192] into WMMA B-fragment table
// layout: [kt(6)][nt(24)][lane(32)][i(16)] bf16, B[k][o] = w_pw[o][k]
// ---------------------------------------------------------------------------
__global__ __launch_bounds__(256) void k_pack_wpw(const float* __restrict__ w_pw,
                                                  __bf16* __restrict__ Bf) {
  int idx = blockIdx.x * 256 + threadIdx.x;
  if (idx >= KT_PW * 24 * 512) return;
  int i    = idx & 15;
  int lane = (idx >> 4) & 31;
  int g    = idx >> 9;            // kt*24 + nt
  int nt = g % 24, kt = g / 24;
  int col = lane & 15;
  int klocal = ((lane >> 4) << 3) + (i & 7) + ((i >> 3) << 4);
  int c = (kt << 5) + klocal;
  int o = (nt << 4) + col;
  Bf[idx] = (__bf16)w_pw[(size_t)o * CIN + c];
}

// ---------------------------------------------------------------------------
// k1: offset conv as im2col GEMM. M=65536 pixels, K=1728, N=32 (18 live).
// One wave per 16-pixel M-tile. Padded-halo xT => branch-free A loads.
// A double-buffered across taps; B staged once per block in LDS (108 KB).
// ---------------------------------------------------------------------------
__global__ __launch_bounds__(256) void k_offset_gemm(const __bf16* __restrict__ xTp,
                                                     const __bf16* __restrict__ Bf,
                                                     const float* __restrict__ b_off,
                                                     float* __restrict__ ws_off) {
  extern __shared__ __bf16 Bs[];            // KT_OFF*2*512 = 55296 elems (108 KB)
  {
    const v8bf* src = (const v8bf*)Bf;
    v8bf* dst = (v8bf*)Bs;
#pragma unroll
    for (int it = 0; it < 27; ++it)         // 6912 chunks / 256 threads
      dst[it * 256 + threadIdx.x] = src[it * 256 + threadIdx.x];
  }
  __syncthreads();

  int wave = (blockIdx.x << 3) + (threadIdx.x >> 5);
  int lane = threadIdx.x & 31;
  int m0 = wave << 4;
  int b = m0 >> 12, h = (m0 >> 6) & 63, w0 = m0 & 63;
  int row   = lane & 15;
  int kbase = (lane >> 4) << 3;

  // padded row base for tap (ki,kj): (b*XPITCH + (h+ki)*66 + w0+row+kj)*CIN
  const __bf16* a00 =
      xTp + ((size_t)(b * XPITCH + h * 66 + w0 + row)) * CIN + kbase;

  FragBF fa[2][6];
#pragma unroll
  for (int ct = 0; ct < 6; ++ct) {          // preload tap kk=0 (ki=0,kj=0)
    fa[0][ct].h[0] = *(const v8bf*)(a00 + ct * 32);
    fa[0][ct].h[1] = *(const v8bf*)(a00 + ct * 32 + 16);
  }

  v8f acc0 = {}, acc1 = {};
#pragma unroll
  for (int kk = 0; kk < 9; ++kk) {
    if (kk < 8) {                           // prefetch next tap's A row
      int kn = kk + 1;
      int ki = kn / 3, kj = kn - ki * 3;    // constant after unroll
      const __bf16* ap = a00 + (size_t)(ki * 66 + kj) * CIN;
#pragma unroll
      for (int ct = 0; ct < 6; ++ct) {
        fa[kn & 1][ct].h[0] = *(const v8bf*)(ap + ct * 32);
        fa[kn & 1][ct].h[1] = *(const v8bf*)(ap + ct * 32 + 16);
      }
    }
#pragma unroll
    for (int ct = 0; ct < 6; ++ct) {
      int kt = kk * 6 + ct;
      const __bf16* bp = Bs + ((size_t)(kt * 2) * 32 + lane) * 16;
      FragBF fb0, fb1;
      fb0.h[0] = *(const v8bf*)bp;          fb0.h[1] = *(const v8bf*)(bp + 8);
      fb1.h[0] = *(const v8bf*)(bp + 512);  fb1.h[1] = *(const v8bf*)(bp + 520);
      acc0 = wmma_bf16(fa[kk & 1][ct].v, fb0.v, acc0);
      acc1 = wmma_bf16(fa[kk & 1][ct].v, fb1.v, acc1);
    }
  }

  int mb  = (lane >> 4) << 3;
  int col = lane & 15;
  float bias0 = b_off[col];
  float bias1 = (col < 2) ? b_off[16 + col] : 0.0f;
#pragma unroll
  for (int r = 0; r < 8; ++r) {
    int pix = m0 + mb + r;
    ws_off[(size_t)pix * 18 + col] = acc0[r] + bias0;
    if (col < 2) ws_off[(size_t)pix * 18 + 16 + col] = acc1[r] + bias1;
  }
}

// ---------------------------------------------------------------------------
// k2: bilinear sampling + depthwise reduce. One wave per pixel; each lane
// owns 6 channels (c = lane + j*32). Padded NHWC => coalesced corner gathers.
// ---------------------------------------------------------------------------
__global__ __launch_bounds__(256) void k_sample_dw(const __bf16* __restrict__ xTp,
                                                   const float* __restrict__ ws_off,
                                                   const float* __restrict__ w_dw,
                                                   __bf16* __restrict__ dwT) {
  int p    = (blockIdx.x << 3) + (threadIdx.x >> 5);
  int lane = threadIdx.x & 31;
  int b = p >> 12, h = (p >> 6) & 63, w = p & 63;
  int bBase = b * XPITCH;
  const float* op = ws_off + (size_t)p * 18;

  float acc[6] = {0.f, 0.f, 0.f, 0.f, 0.f, 0.f};
#pragma unroll
  for (int k = 0; k < 9; ++k) {
    int ki = k / 3, kj = k - ki * 3;
    float py = (float)(h + ki - 1) + op[2 * k];
    float px = (float)(w + kj - 1) + op[2 * k + 1];
    float y0f = floorf(py), x0f = floorf(px);
    float dy = py - y0f, dx = px - x0f;
    int y0 = (int)y0f, x0 = (int)x0f;
    float wc[4] = { (1.f - dy) * (1.f - dx), (1.f - dy) * dx,
                    dy * (1.f - dx),         dy * dx };
    int yy[4] = { y0, y0, y0 + 1, y0 + 1 };
    int xc[4] = { x0, x0 + 1, x0, x0 + 1 };
    float samp[6] = {0.f, 0.f, 0.f, 0.f, 0.f, 0.f};
#pragma unroll
    for (int cnr = 0; cnr < 4; ++cnr) {
      if ((unsigned)yy[cnr] < 64u && (unsigned)xc[cnr] < 64u) {
        const __bf16* bp =
            xTp + (size_t)(bBase + (yy[cnr] + 1) * 66 + xc[cnr] + 1) * CIN + lane;
        float wgt = wc[cnr];
#pragma unroll
        for (int j = 0; j < 6; ++j) samp[j] += wgt * (float)bp[j * 32];
      }
    }
#pragma unroll
    for (int j = 0; j < 6; ++j) acc[j] += samp[j] * w_dw[(lane + j * 32) * 9 + k];
  }
  __bf16* o = dwT + (size_t)p * CIN + lane;
#pragma unroll
  for (int j = 0; j < 6; ++j) o[j * 32] = (__bf16)acc[j];
}

// ---------------------------------------------------------------------------
// k3: pointwise GEMM. M=65536, K=192, N=384. Block = 8 waves sharing one
// N-group (64 cols): its 24 KB B-slice staged in LDS (8x reuse). A fragments
// double-buffered across K-tiles. Output scattered to f32 NCHW.
// ---------------------------------------------------------------------------
__global__ __launch_bounds__(256) void k_pw_gemm(const __bf16* __restrict__ dwT,
                                                 const __bf16* __restrict__ Bf,
                                                 float* __restrict__ out) {
  __shared__ __bf16 Bs[KT_PW * 4 * 512];    // 12288 elems = 24 KB
  int ng   = blockIdx.x % 6;
  int mgrp = blockIdx.x / 6;
  {
    // 1536 v8bf chunks; chunk i = [kt][t][lane][half]
#pragma unroll
    for (int it = 0; it < 6; ++it) {
      int i = it * 256 + threadIdx.x;
      int hh    = i & 1;
      int laneq = (i >> 1) & 31;
      int tq    = (i >> 6) & 3;
      int ktq   = i >> 8;
      ((v8bf*)Bs)[i] =
          ((const v8bf*)Bf)[(((ktq * 24 + ng * 4 + tq) * 32 + laneq) << 1) + hh];
    }
  }
  __syncthreads();

  int wave = threadIdx.x >> 5;
  int lane = threadIdx.x & 31;
  int m0 = ((mgrp << 3) + wave) << 4;
  int row   = lane & 15;
  int kbase = (lane >> 4) << 3;

  const __bf16* abase = dwT + (size_t)(m0 + row) * CIN + kbase;
  FragBF fa[2];
  fa[0].h[0] = *(const v8bf*)abase;
  fa[0].h[1] = *(const v8bf*)(abase + 16);

  v8f acc[4];
#pragma unroll
  for (int t = 0; t < 4; ++t)
#pragma unroll
    for (int e = 0; e < 8; ++e) acc[t][e] = 0.0f;

#pragma unroll
  for (int kt = 0; kt < KT_PW; ++kt) {
    if (kt < KT_PW - 1) {
      const __bf16* ap = abase + ((kt + 1) << 5);
      fa[(kt + 1) & 1].h[0] = *(const v8bf*)ap;
      fa[(kt + 1) & 1].h[1] = *(const v8bf*)(ap + 16);
    }
#pragma unroll
    for (int t = 0; t < 4; ++t) {
      const __bf16* bp = Bs + ((size_t)((kt * 4 + t) * 32 + lane)) * 16;
      FragBF fb;
      fb.h[0] = *(const v8bf*)bp;
      fb.h[1] = *(const v8bf*)(bp + 8);
      acc[t] = wmma_bf16(fa[kt & 1].v, fb.v, acc[t]);
    }
  }

  int mb  = (lane >> 4) << 3;
  int col = lane & 15;
#pragma unroll
  for (int t = 0; t < 4; ++t) {
    int o = ((ng * 4 + t) << 4) + col;
#pragma unroll
    for (int r = 0; r < 8; ++r) {
      int m = m0 + mb + r;
      int bb = m >> 12, hw = m & 4095;
      out[((size_t)(bb * COUT + o) << 12) + hw] = acc[t][r];
    }
  }
}

// ---------------------------------------------------------------------------
extern "C" void kernel_launch(void* const* d_in, const int* in_sizes, int n_in,
                              void* d_out, int out_size, void* d_ws, size_t ws_size,
                              hipStream_t stream) {
  (void)in_sizes; (void)n_in; (void)out_size; (void)ws_size;
  const float* x     = (const float*)d_in[0];
  const float* w_off = (const float*)d_in[1];
  const float* b_off = (const float*)d_in[2];
  const float* w_dw  = (const float*)d_in[3];
  const float* w_pw  = (const float*)d_in[4];
  float* out = (float*)d_out;

  char* ws = (char*)d_ws;
  __bf16* xTp    = (__bf16*)(ws);                    // 26,763,264 B bf16 padded NHWC x
  float*  ws_off = (float*) (ws + 26763264);         //  4,718,592 B [pix][18] offsets
  __bf16* dwT    = (__bf16*)(ws + 31481856);         // 25,165,824 B bf16 NHWC out_dw
  __bf16* BfO    = (__bf16*)(ws + 56647680);         //    110,592 B packed w_off frags
  __bf16* BfP    = (__bf16*)(ws + 56758272);         //    147,456 B packed w_pw frags

  int nchunks = (BATCH * XPITCH * CIN * 2) / 16;     // 1,672,704
  k_zero16    <<<(nchunks + 255) / 256, 256, 0, stream>>>((v8bf*)xTp, nchunks);
  k_x_to_nhwc <<<16 * 6 * 128, 256, 0, stream>>>(x, xTp);
  k_pack_woff <<<(KT_OFF * 2 * 512 + 255) / 256, 256, 0, stream>>>(w_off, BfO);
  k_pack_wpw  <<<(KT_PW * 24 * 512 + 255) / 256, 256, 0, stream>>>(w_pw, BfP);
  k_offset_gemm<<<NPIX / 16 / 8, 256, KT_OFF * 2 * 512 * sizeof(__bf16), stream>>>(
      xTp, BfO, b_off, ws_off);
  k_sample_dw <<<NPIX / 8, 256, 0, stream>>>(xTp, ws_off, w_dw, dwT);
  k_pw_gemm   <<<(NPIX / 16 / 8) * 6, 256, 0, stream>>>(dwT, BfP, out);
}
